// PoincareCritic_35888746725469
// MI455X (gfx1250) — matrix operand order repaired and checked by reading, not compiled, not run on hardware
//
#include <hip/hip_runtime.h>
#include <hip/hip_bf16.h>
#include <math.h>

typedef __attribute__((ext_vector_type(16))) _Float16 v16h;
typedef __attribute__((ext_vector_type(8)))  float    v8f;

#define BATCH            32768
#define KDIM             1472
#define LATENT           64
#define NBINS            51
#define WAVES_PER_BLOCK  4
#define ROWS_PER_WAVE    32      // two 16-row WMMA tiles per wave
#define ROWS_PER_BLOCK   128     // 4 waves * 32 rows
#define XSTRIDE          68      // padded LDS row stride (16B aligned)
#define PFDIST           256     // prefetch distance in floats (8 K-steps = 1KB/row)
#define KLAST            (KDIM - 32)

#define RHO_MAX   0.95f
#define EPS_H     1e-15f
#define TANH_CLIP (1.0f - 1e-5f)
#define F32_EPS   1.1920929e-7f

static_assert(KDIM % 64 == 0, "K must be a multiple of 64 for the 2x-unrolled WMMA loop");

// Raw 32-float K-slice of one A row-fragment (two 32B chunks per lane).
struct ARaw { float4 c0, c1, c2, c3; };

__device__ __forceinline__ ARaw load_a_raw(const float* __restrict__ p) {
  ARaw r;
  r.c0 = *(const float4*)(p + 0);
  r.c1 = *(const float4*)(p + 4);
  r.c2 = *(const float4*)(p + 16);
  r.c3 = *(const float4*)(p + 20);
  return r;
}

// A fragment (16x32 f16, MxK): lane<16 holds row M=lane, K-offsets {0..7,16..23};
// lane>=16 holds M=lane-16, K-offsets {8..15,24..31}. Pointer pre-offset by 8*hi.
__device__ __forceinline__ v16h cvt_a(const ARaw& r) {
  v16h o;
  o[0]=(_Float16)r.c0.x;  o[1]=(_Float16)r.c0.y;  o[2]=(_Float16)r.c0.z;  o[3]=(_Float16)r.c0.w;
  o[4]=(_Float16)r.c1.x;  o[5]=(_Float16)r.c1.y;  o[6]=(_Float16)r.c1.z;  o[7]=(_Float16)r.c1.w;
  o[8]=(_Float16)r.c2.x;  o[9]=(_Float16)r.c2.y;  o[10]=(_Float16)r.c2.z; o[11]=(_Float16)r.c2.w;
  o[12]=(_Float16)r.c3.x; o[13]=(_Float16)r.c3.y; o[14]=(_Float16)r.c3.z; o[15]=(_Float16)r.c3.w;
  return o;
}

// B fragment (32x16 f16, KxN): lane<16 holds column N=lane, K={0..7,16..23}; lane>=16 holds
// N=lane-16, K={8..15,24..31}. B = W^T, so a B column is a contiguous W row. Pointer pre-offset by 8*hi.
__device__ __forceinline__ v16h cvt_b_frag(const float* __restrict__ p) {
  float4 c0 = *(const float4*)(p + 0);
  float4 c1 = *(const float4*)(p + 4);
  float4 c2 = *(const float4*)(p + 16);
  float4 c3 = *(const float4*)(p + 20);
  v16h o;
  o[0]=(_Float16)c0.x;  o[1]=(_Float16)c0.y;  o[2]=(_Float16)c0.z;  o[3]=(_Float16)c0.w;
  o[4]=(_Float16)c1.x;  o[5]=(_Float16)c1.y;  o[6]=(_Float16)c1.z;  o[7]=(_Float16)c1.w;
  o[8]=(_Float16)c2.x;  o[9]=(_Float16)c2.y;  o[10]=(_Float16)c2.z; o[11]=(_Float16)c2.w;
  o[12]=(_Float16)c3.x; o[13]=(_Float16)c3.y; o[14]=(_Float16)c3.z; o[15]=(_Float16)c3.w;
  return o;
}

#define WMMA_F16(A, B, C) \
  __builtin_amdgcn_wmma_f32_16x16x32_f16(false, (A), false, (B), (short)0, (C), false, false)

// One 32-wide K-step: 4 B fragments (demand, L2-hot) x 2 A fragments -> 8 WMMAs.
#define GEMM_HALF(RA0, RA1, KOFF) do {                 \
    v16h b0 = cvt_b_frag(wPtr0 + (KOFF));              \
    v16h b1 = cvt_b_frag(wPtr1 + (KOFF));              \
    v16h b2 = cvt_b_frag(wPtr2 + (KOFF));              \
    v16h b3 = cvt_b_frag(wPtr3 + (KOFF));              \
    v16h f0 = cvt_a(RA0);                              \
    v16h f1 = cvt_a(RA1);                              \
    a00 = WMMA_F16(f0, b0, a00);                       \
    a01 = WMMA_F16(f0, b1, a01);                       \
    a02 = WMMA_F16(f0, b2, a02);                       \
    a03 = WMMA_F16(f0, b3, a03);                       \
    a10 = WMMA_F16(f1, b0, a10);                       \
    a11 = WMMA_F16(f1, b1, a11);                       \
    a12 = WMMA_F16(f1, b2, a12);                       \
    a13 = WMMA_F16(f1, b3, a13);                       \
  } while (0)

__global__ __launch_bounds__(128)
void poincare_critic_fused(const float* __restrict__ A,      // (32768,1472)
                           const float* __restrict__ W,      // (64,1472)
                           const float* __restrict__ bias,   // (64)
                           const float* __restrict__ rms_w,  // (64)
                           const float* __restrict__ xi_p,   // scalar
                           const float* __restrict__ z,      // (64,51)
                           const float* __restrict__ r,      // (51)
                           const float* __restrict__ support,// (51)
                           float* __restrict__ out_value,    // (32768)
                           float* __restrict__ out_logits,   // (32768,51)
                           float* __restrict__ out_probs) {  // (32768,51)
  __shared__ float s_x[ROWS_PER_BLOCK][XSTRIDE];   // GEMM result, later reused for w/e
  __shared__ float s_zt[NBINS][LATENT];            // z transposed: s_zt[j][k] = z[k][j]
  __shared__ float s_zn[NBINS], s_c2r[NBINS], s_s2r[NBINS], s_sup[NBINS];
  __shared__ float s_bias[LATENT], s_rmsw[LATENT];

  const int tid  = threadIdx.x;
  const int wave = tid >> 5;
  const int lane = tid & 31;
  const int lh   = lane & 15;
  const int hi   = lane >> 4;    // 0 or 1

  // ---- stage small operands (epilogue constants) ----
  if (tid < NBINS) {
    float ss = 0.f;
    for (int k = 0; k < LATENT; ++k) {
      float v = z[k * NBINS + tid];
      s_zt[tid][k] = v;
      ss += v * v;
    }
    s_zn[tid]  = fmaxf(sqrtf(ss), EPS_H);
    float tr   = 2.0f * r[tid];
    s_c2r[tid] = coshf(tr);
    s_s2r[tid] = sinhf(tr);
    s_sup[tid] = support[tid];
  }
  if (tid >= 64 && tid < 128) {
    int i = tid - 64;
    s_bias[i] = bias[i];
    s_rmsw[i] = rms_w[i];
  }
  __syncthreads();

  // ---- GEMM: each wave computes a 32x64 tile via 8 accumulators ----
  const int rowBase = blockIdx.x * ROWS_PER_BLOCK + wave * ROWS_PER_WAVE;

  const float* aPtr0 = A + (size_t)(rowBase + 0  + lh) * KDIM + 8 * hi;
  const float* aPtr1 = A + (size_t)(rowBase + 16 + lh) * KDIM + 8 * hi;
  const float* wPtr0 = W + (size_t)( 0 + lh) * KDIM + 8 * hi;
  const float* wPtr1 = W + (size_t)(16 + lh) * KDIM + 8 * hi;
  const float* wPtr2 = W + (size_t)(32 + lh) * KDIM + 8 * hi;
  const float* wPtr3 = W + (size_t)(48 + lh) * KDIM + 8 * hi;

  const v8f z8 = {0.f, 0.f, 0.f, 0.f, 0.f, 0.f, 0.f, 0.f};
  v8f a00 = z8, a01 = z8, a02 = z8, a03 = z8;
  v8f a10 = z8, a11 = z8, a12 = z8, a13 = z8;

  // Prime ping buffer (HBM stream).
  ARaw ra0 = load_a_raw(aPtr0);
  ARaw ra1 = load_a_raw(aPtr1);

  // 2x-unrolled ping-pong: no buffer copies, branch-free tail via clamped offsets.
  for (int k0 = 0; k0 < KDIM; k0 += 64) {
    // Fire-and-forget prefetch of future A/W cachelines (global_prefetch_b8).
    int kp = k0 + PFDIST; if (kp > KLAST) kp = KLAST;   // uniform s_cselect, no branch
    __builtin_prefetch(aPtr0 + kp, 0, 3);
    __builtin_prefetch(aPtr1 + kp, 0, 3);
    __builtin_prefetch(wPtr0 + kp, 0, 3);
    __builtin_prefetch(wPtr1 + kp, 0, 3);
    __builtin_prefetch(wPtr2 + kp, 0, 3);
    __builtin_prefetch(wPtr3 + kp, 0, 3);

    const int ka = k0 + 32;                              // always < KDIM (46 steps, even)
    ARaw rb0 = load_a_raw(aPtr0 + ka);                   // pong loads in flight ...
    ARaw rb1 = load_a_raw(aPtr1 + ka);
    GEMM_HALF(ra0, ra1, k0);                             // ... while ping computes

    int kb = k0 + 64; if (kb > KLAST) kb = KLAST;        // tail: redundant reload, no branch
    ra0 = load_a_raw(aPtr0 + kb);                        // ping loads in flight ...
    ra1 = load_a_raw(aPtr1 + kb);
    GEMM_HALF(rb0, rb1, ka);                             // ... while pong computes
  }

  // ---- spill x (+bias) to LDS. C layout: VGPR e -> M = e (lanes 0-15) / 8+e (lanes 16-31), N = lh ----
  {
    const int rl0 = wave * ROWS_PER_WAVE + hi * 8;
#pragma unroll
    for (int e = 0; e < 8; ++e) {
      int r0 = rl0 + e, r1 = rl0 + e + 16;
      s_x[r0][ 0 + lh] = a00[e] + s_bias[ 0 + lh];
      s_x[r0][16 + lh] = a01[e] + s_bias[16 + lh];
      s_x[r0][32 + lh] = a02[e] + s_bias[32 + lh];
      s_x[r0][48 + lh] = a03[e] + s_bias[48 + lh];
      s_x[r1][ 0 + lh] = a10[e] + s_bias[ 0 + lh];
      s_x[r1][16 + lh] = a11[e] + s_bias[16 + lh];
      s_x[r1][32 + lh] = a12[e] + s_bias[32 + lh];
      s_x[r1][48 + lh] = a13[e] + s_bias[48 + lh];
    }
  }
  __syncthreads();

  // ---- hyperbolic epilogue: one thread per row ----
  const int row  = tid;                              // 128 rows, 128 threads
  const int gRow = blockIdx.x * ROWS_PER_BLOCK + row;
  const float xiv = xi_p[0];
  const float sig = RHO_MAX / (1.0f + __expf(-xiv)); // RHO_MAX * sigmoid(xi)

  float xv[LATENT];
  float ss = 0.f;
#pragma unroll
  for (int k4 = 0; k4 < LATENT / 4; ++k4) {
    float4 c = *(const float4*)(&s_x[row][4 * k4]);
    xv[4*k4+0] = c.x; xv[4*k4+1] = c.y; xv[4*k4+2] = c.z; xv[4*k4+3] = c.w;
    ss += c.x*c.x + c.y*c.y + c.z*c.z + c.w*c.w;
  }
  // RMSNorm + learned scale + sigmoid scale
  const float inv = rsqrtf(ss * (1.0f / LATENT) + F32_EPS);
  float n2 = 0.f;
#pragma unroll
  for (int k = 0; k < LATENT; ++k) {
    float v = xv[k] * inv * s_rmsw[k] * sig;
    xv[k] = v;
    n2 += v * v;
  }
  // expmap at origin
  const float xn = fmaxf(sqrtf(n2), EPS_H);
  const float th = tanhf(xn);
  const float sc = th / xn;
#pragma unroll
  for (int k = 0; k < LATENT; ++k) xv[k] *= sc;      // x_h, ||x_h|| = th
  const float lam = 2.0f / (1.0f - th * th);

  // Poincare FC: pass 1 -> w_j, stash in s_x row (x no longer needed there)
  float sw2 = 0.f;
  for (int j = 0; j < NBINS; ++j) {
    float s = 0.f;
#pragma unroll
    for (int k4 = 0; k4 < LATENT / 4; ++k4) {
      float4 c = *(const float4*)(&s_zt[j][4 * k4]);   // broadcast across threads
      s += xv[4*k4+0]*c.x + xv[4*k4+1]*c.y + xv[4*k4+2]*c.z + xv[4*k4+3]*c.w;
    }
    float zn    = s_zn[j];
    float inner = s / zn;
    float arg   = lam * inner * s_c2r[j] - (lam - 1.0f) * s_s2r[j];
    float v     = 2.0f * zn * asinhf(arg);
    float w     = sinhf(v);
    sw2 += w * w;
    s_x[row][j] = w;
  }

  // logmap at origin -> logits (+ softmax max)
  const float denom = 1.0f + sqrtf(1.0f + sw2);
  const float yn    = sqrtf(sw2) / denom;            // ||y||
  const float ync   = fmaxf(yn, EPS_H);
  const float at    = atanhf(fminf(ync, TANH_CLIP));
  const float lf    = (at / ync) / denom;            // logits = lf * w

  float m = -INFINITY;
  for (int j = 0; j < NBINS; ++j) {
    float lg = lf * s_x[row][j];
    s_x[row][j] = lg;
    out_logits[(size_t)gRow * NBINS + j] = lg;
    m = fmaxf(m, lg);
  }
  float se = 0.f;
  for (int j = 0; j < NBINS; ++j) {
    float e = __expf(s_x[row][j] - m);
    s_x[row][j] = e;
    se += e;
  }
  const float rse = 1.0f / se;
  float val = 0.f;
  for (int j = 0; j < NBINS; ++j) {
    float p = s_x[row][j] * rse;
    out_probs[(size_t)gRow * NBINS + j] = p;
    val += p * s_sup[j];
  }
  out_value[gRow] = val;
}

extern "C" void kernel_launch(void* const* d_in, const int* in_sizes, int n_in,
                              void* d_out, int out_size, void* d_ws, size_t ws_size,
                              hipStream_t stream) {
  (void)in_sizes; (void)n_in; (void)d_ws; (void)ws_size; (void)out_size;
  const float* A       = (const float*)d_in[0]; // encoder_out
  const float* W       = (const float*)d_in[1];
  const float* bias    = (const float*)d_in[2];
  const float* rms_w   = (const float*)d_in[3];
  const float* xi      = (const float*)d_in[4];
  const float* z       = (const float*)d_in[5];
  const float* r       = (const float*)d_in[6];
  const float* support = (const float*)d_in[7];

  float* out        = (float*)d_out;
  float* out_value  = out;                          // (32768,)
  float* out_logits = out + BATCH;                  // (32768,51)
  float* out_probs  = out + BATCH + BATCH * NBINS;  // (32768,51)

  dim3 grid(BATCH / ROWS_PER_BLOCK);                // 256 blocks
  dim3 block(ROWS_PER_BLOCK);                       // 128 threads = 4 waves
  poincare_critic_fused<<<grid, block, 0, stream>>>(
      A, W, bias, rms_w, xi, z, r, support, out_value, out_logits, out_probs);
}
